// BiRNNLMwithDropout_30262339568170
// MI455X (gfx1250) — matrix operand order repaired and checked by reading, not compiled, not run on hardware
//
#include <hip/hip_runtime.h>
#include <hip/hip_bf16.h>

typedef _Float16 v16h __attribute__((ext_vector_type(16)));
typedef float    v8f  __attribute__((ext_vector_type(8)));

#define V_   32000
#define S_   256
#define B_   16
#define E_   32
#define H_   16
#define NT   (V_ / 16)          /* 2000 N-tiles */
#define NWAVE 16                /* waves per block in k_gemm_lsm */
#define NIT  (NT / NWAVE)       /* 125 uniform iterations per wave */
#define KEEP_INV (1.0f / 0.6f)

/* ---- workspace layout (bytes) ---- */
#define WSB_OFF   0u                         /* f16 B tiles: NT*32*16*2 = 2,048,000 */
#define WSA_OFF   2097152u                   /* f16 A tiles: 256*32*16*2 = 262,144  */
#define XPROJ_OFF (WSA_OFF + 262144u)        /* f32 [2][S][B][H] = 524,288          */
#define STATE_OFF (XPROJ_OFF + 524288u)      /* f32 [2][S][B][H] = 524,288          */

/* ------------------------------------------------------------------ */
/* Kernel 1: repack W_ho (f32, [2H=32, V] row-major) into f16 B-tiles */
/* B VGPR layout (16-bit 32x16): lane = 16*((k/8)&1) + n, half = 8*(k>=16) + k%8.  */
/* Stored per tile t as wsB[(t*32 + lane)*16 + half].                 */
/* ------------------------------------------------------------------ */
__global__ __launch_bounds__(256) void k_pack_B(const float* __restrict__ W_ho,
                                                _Float16* __restrict__ wsB) {
    int v = blockIdx.x * 256 + threadIdx.x;
    if (v >= V_) return;
    int t = v >> 4, n = v & 15;
    _Float16* d0 = wsB + ((size_t)(t * 32 + n) * 16);        /* K = 0..7, 16..23  */
    _Float16* d1 = wsB + ((size_t)(t * 32 + n + 16) * 16);   /* K = 8..15, 24..31 */
#pragma unroll
    for (int k = 0; k < 16; ++k) d0[k] = (_Float16)W_ho[(size_t)((k < 8) ? k : k + 8) * V_ + v];
#pragma unroll
    for (int k = 0; k < 16; ++k) d1[k] = (_Float16)W_ho[(size_t)((k < 8) ? k + 8 : k + 16) * V_ + v];
}

/* ------------------------------------------------------------------ */
/* Kernel 2: x_proj[dir][s][b][j] = b_ih[j] + emb[tok[s][b]] @ W_ih[:E] */
/* ------------------------------------------------------------------ */
__global__ __launch_bounds__(256) void k_xproj(const int* __restrict__ tokens,
                                               const float* __restrict__ emb,
                                               const float* __restrict__ W_lr,
                                               const float* __restrict__ b_lr,
                                               const float* __restrict__ W_rl,
                                               const float* __restrict__ b_rl,
                                               float* __restrict__ xproj) {
    int idx = blockIdx.x * 256 + threadIdx.x;   /* 2*S*B*H = 131072 */
    int j = idx & 15;
    int b = (idx >> 4) & 15;
    int s = (idx >> 8) & 255;
    int dir = idx >> 16;
    const float* W  = dir ? W_rl : W_lr;
    const float* bi = dir ? b_rl : b_lr;
    int tok = tokens[s * B_ + b];
    const float* erow = emb + (size_t)tok * E_;
    float acc = bi[j];
#pragma unroll 8
    for (int e = 0; e < E_; ++e) acc += erow[e] * W[e * H_ + j];
    xproj[idx] = acc;
}

/* ------------------------------------------------------------------ */
/* Kernel 3: the two serial scans. blockIdx.x = direction.            */
/* h_new = tanh(xproj + h_prev @ W_hh) * mask * (1/KEEP)              */
/* ------------------------------------------------------------------ */
__global__ __launch_bounds__(256) void k_rnn(const float* __restrict__ xproj,
                                             const float* __restrict__ W_lr,
                                             const float* __restrict__ W_rl,
                                             const float* __restrict__ mask_lr,
                                             const float* __restrict__ mask_rl,
                                             const float* __restrict__ h0,
                                             float* __restrict__ states) {
    int dir = blockIdx.x;
    int tid = threadIdx.x;
    int b = tid >> 4, j = tid & 15;
    const float* W    = dir ? W_rl : W_lr;        /* rows E..E+H are W_hh */
    const float* mask = dir ? mask_rl : mask_lr;
    const float* xp   = xproj + (size_t)dir * S_ * B_ * H_;
    float* st         = states + (size_t)dir * S_ * B_ * H_;

    __shared__ float Wh[16][16];
    __shared__ float hbuf[2][16][17];
    Wh[b][j] = W[(E_ + b) * H_ + j];              /* reuse b as k index  */
    hbuf[0][b][j] = h0[j];
    __syncthreads();

    int cur = 0;
    for (int step = 0; step < S_; ++step) {
        int s = dir ? (S_ - 1 - step) : step;
        float acc = xp[((size_t)s * B_ + b) * H_ + j];
#pragma unroll
        for (int k = 0; k < H_; ++k) acc += hbuf[cur][b][k] * Wh[k][j];
        float hn = tanhf(acc) * mask[((size_t)s * B_ + b) * H_ + j] * KEEP_INV;
        hbuf[cur ^ 1][b][j] = hn;
        st[((size_t)s * B_ + b) * H_ + j] = hn;
        cur ^= 1;
        __syncthreads();
    }
}

/* ------------------------------------------------------------------ */
/* Kernel 4: build f16 A-tiles (16x32, interleaved 16-bit A layout):  */
/* lanes 0-15 (M=lane): halves 0-7 = K0-7,  halves 8-15 = K16-23      */
/* lanes 16-31(M=l-16): halves 0-7 = K8-15, halves 8-15 = K24-31      */
/* hcat K<16 = hLR_used (s==0? h0 : s_lr[s-1]),                       */
/*      K>=16= hRL_used (s==S-1? h0 : s_rl[s+1])                      */
/* ------------------------------------------------------------------ */
__global__ void k_pack_A(const float* __restrict__ states,
                         const float* __restrict__ h0,
                         _Float16* __restrict__ wsA) {
    int s = blockIdx.x;
    int l = threadIdx.x;          /* 0..31 */
    int M = l & 15;
    int k0 = (l < 16) ? 0 : 8;
    const float* sLR = states;
    const float* sRL = states + (size_t)S_ * B_ * H_;
    _Float16* dst = wsA + (size_t)(s * 32 + l) * 16;
#pragma unroll
    for (int i = 0; i < 8; ++i) {
        float vLR = (s == 0) ? h0[k0 + i]
                             : sLR[(((size_t)(s - 1)) * B_ + M) * H_ + k0 + i];
        dst[i] = (_Float16)vLR;
    }
#pragma unroll
    for (int i = 0; i < 8; ++i) {
        float vRL = (s == S_ - 1) ? h0[k0 + i]
                                  : sRL[(((size_t)(s + 1)) * B_ + M) * H_ + k0 + i];
        dst[8 + i] = (_Float16)vRL;
    }
}

/* ------------------------------------------------------------------ */
/* Kernel 5: fused logits GEMM (WMMA f32_16x16x32_f16) + log_softmax. */
/* One block per s (16-row M-tile), 16 waves; each wave runs exactly  */
/* NIT=125 uniform iterations (no divergent backedge / exec masking). */
/* Pass 1 accumulates per-row sum(exp(x)); pass 2 recomputes & stores */
/* x - log(sum). C/D layout: each lane's 8 acc regs share one column. */
/* ------------------------------------------------------------------ */
__global__ __launch_bounds__(512) void k_gemm_lsm(const _Float16* __restrict__ wsA,
                                                  const _Float16* __restrict__ wsB,
                                                  const float* __restrict__ b_ho,
                                                  float* __restrict__ out) {
    int s    = blockIdx.x;
    int tid  = threadIdx.x;
    int lane = tid & 31;
    int wave = tid >> 5;          /* 0..15 */
    int col  = lane & 15;
    int rbase = (lane < 16) ? 0 : 8;

    v16h a = *(const v16h*)(wsA + (size_t)(s * 32 + lane) * 16);

    float rsum[8];
#pragma unroll
    for (int i = 0; i < 8; ++i) rsum[i] = 0.0f;

    /* ---- pass 1: row sums of exp(logit) ---- */
    for (int it = 0; it < NIT; ++it) {
        int t = wave + (it << 4);
        v16h bm = *(const v16h*)(wsB + (size_t)(t * 32 + lane) * 16);
        v8f c = {};
        c = __builtin_amdgcn_wmma_f32_16x16x32_f16(false, a, false, bm,
                                                   (short)0, c, false, false);
        float bias = b_ho[t * 16 + col];
#pragma unroll
        for (int i = 0; i < 8; ++i) rsum[i] += __expf(c[i] + bias);
    }

    /* reduce across the 16 columns of each half-wave */
#pragma unroll
    for (int m = 1; m <= 8; m <<= 1) {
#pragma unroll
        for (int i = 0; i < 8; ++i) rsum[i] += __shfl_xor(rsum[i], m, 32);
    }

    __shared__ float lsum[16][16];   /* [wave][row] */
    __shared__ float logz[16];
    if ((lane & 15) == 0) {          /* lanes 0 and 16 */
#pragma unroll
        for (int i = 0; i < 8; ++i) lsum[wave][rbase + i] = rsum[i];
    }
    __syncthreads();
    if (tid < 16) {
        float tot = 0.0f;
#pragma unroll
        for (int w = 0; w < 16; ++w) tot += lsum[w][tid];
        logz[tid] = __logf(tot);
    }
    __syncthreads();

    float lz[8];
#pragma unroll
    for (int i = 0; i < 8; ++i) lz[i] = logz[rbase + i];

    /* ---- pass 2: recompute logits, write log-softmax ---- */
    for (int it = 0; it < NIT; ++it) {
        int t = wave + (it << 4);
        v16h bm = *(const v16h*)(wsB + (size_t)(t * 32 + lane) * 16);
        v8f c = {};
        c = __builtin_amdgcn_wmma_f32_16x16x32_f16(false, a, false, bm,
                                                   (short)0, c, false, false);
        float bias = b_ho[t * 16 + col];
#pragma unroll
        for (int i = 0; i < 8; ++i) {
            int row = s * 16 + rbase + i;
            out[(size_t)row * V_ + t * 16 + col] = c[i] + bias - lz[i];
        }
    }
}

/* ------------------------------------------------------------------ */
extern "C" void kernel_launch(void* const* d_in, const int* in_sizes, int n_in,
                              void* d_out, int out_size, void* d_ws, size_t ws_size,
                              hipStream_t stream) {
    const int*   tokens  = (const int*)d_in[0];
    const float* mask_lr = (const float*)d_in[1];
    const float* mask_rl = (const float*)d_in[2];
    const float* emb     = (const float*)d_in[3];
    const float* W_lr    = (const float*)d_in[4];
    const float* b_lr    = (const float*)d_in[5];
    const float* W_rl    = (const float*)d_in[6];
    const float* b_rl    = (const float*)d_in[7];
    const float* W_ho    = (const float*)d_in[8];
    const float* b_ho    = (const float*)d_in[9];
    const float* h0      = (const float*)d_in[10];
    float* out = (float*)d_out;

    char* ws = (char*)d_ws;
    _Float16* wsB   = (_Float16*)(ws + WSB_OFF);
    _Float16* wsA   = (_Float16*)(ws + WSA_OFF);
    float*    xproj = (float*)(ws + XPROJ_OFF);
    float*    states= (float*)(ws + STATE_OFF);

    k_pack_B<<<(V_ + 255) / 256, 256, 0, stream>>>(W_ho, wsB);
    k_xproj<<<(2 * S_ * B_ * H_) / 256, 256, 0, stream>>>(tokens, emb, W_lr, b_lr,
                                                          W_rl, b_rl, xproj);
    k_rnn<<<2, 256, 0, stream>>>(xproj, W_lr, W_rl, mask_lr, mask_rl, h0, states);
    k_pack_A<<<S_, 32, 0, stream>>>(states, h0, wsA);
    k_gemm_lsm<<<S_, 512, 0, stream>>>(wsA, wsB, b_ho, out);
}
